// BioFormerModel_87591563034696
// MI455X (gfx1250) — compile-verified, matrix operand order checked by reading, not run on hardware
//
#include <hip/hip_runtime.h>
#include <stdint.h>
#include <stddef.h>

// ---------------------------------------------------------------------------
// BioFormer forward for MI455X (gfx1250, wave32, WMMA).
// All large GEMMs use v_wmma_f32_16x16x32_bf16 (bf16 A/B, f32 accumulate).
// Attention scores / z-LayerNorm / softmax are fused VALU streaming kernels
// (they are HBM-bound on the 256MB z tensor, not compute-bound).
// ---------------------------------------------------------------------------

typedef __attribute__((ext_vector_type(16))) __bf16 v16bf;
typedef __attribute__((ext_vector_type(8)))  __bf16 v8bf;
typedef __attribute__((ext_vector_type(8)))  float  v8f;

static constexpr int kR  = 1024;   // rows (tokens)
static constexpr int kCM = 512;    // c_m
static constexpr int kCZ = 64;     // c_z
static constexpr int kCH = 64;     // c_hidden (total qkv width)
static constexpr int kNH = 8;      // heads
static constexpr int kDH = 8;      // per-head dim
static constexpr int kFF = 2048;   // transition hidden

#define GF_RELU  1
#define GF_LEAKY 2
#define GF_RESID 4

// ---------------------------------------------------------------------------
// Generic bf16 WMMA GEMM:  C[m,n] = sum_k A[m,k]*Bt[n,k] (+bias[n])
// optional relu / leaky-relu / residual-add, optional bf16 mirror output.
// One 16x16 output tile per wave; K consumed 32 at a time.
// Batched via blockIdx.z with element strides sA/sB/sCin/sC.
// ---------------------------------------------------------------------------
__global__ __launch_bounds__(256)
void gemm_bf16_wmma(const __bf16* __restrict__ A,  long long sA,
                    const __bf16* __restrict__ Bt, long long sB,
                    const float*  __restrict__ bias,
                    const float*  __restrict__ Cin, long long sCin,
                    float*        __restrict__ C,   long long sC,
                    __bf16*       __restrict__ Cb,
                    int M, int N, int K, int lda, int ldb, int ldc, int flags)
{
  const int wave   = threadIdx.x >> 5;
  const int lane   = threadIdx.x & 31;
  const int tilesN = N >> 4;
  const int tilesT = (M >> 4) * tilesN;
  const int tile   = blockIdx.x * 8 + wave;
  if (tile >= tilesT) return;                 // wave-uniform: EXEC stays all-1s

  const size_t bz = blockIdx.z;
  const __bf16* Ab = A  + bz * (size_t)sA;
  const __bf16* Bb = Bt + bz * (size_t)sB;
  float*        Cc = C  + bz * (size_t)sC;
  const float*  Ci = Cin ? (Cin + bz * (size_t)sCin) : nullptr;

  const int tm = tile / tilesN;
  const int tn = tile - tm * tilesN;
  const int hi = lane >> 4;                   // upper-half lanes carry K+8 / M+8
  const int lr = lane & 15;

  // A fragment: row = tm*16+lr, 8 contiguous bf16 at k0+hi*8 and k0+16+hi*8
  const __bf16* ap = Ab + (size_t)(tm * 16 + lr) * lda + hi * 8;
  // B fragment (Bt is B^T, [N,K] row-major): col = tn*16+lr, same K pattern
  const __bf16* bp = Bb + (size_t)(tn * 16 + lr) * ldb + hi * 8;

  v8f acc = {0.f, 0.f, 0.f, 0.f, 0.f, 0.f, 0.f, 0.f};
  for (int k0 = 0; k0 < K; k0 += 32) {
    v8bf alo = *(const v8bf*)(ap + k0);
    v8bf ahi = *(const v8bf*)(ap + k0 + 16);
    v8bf blo = *(const v8bf*)(bp + k0);
    v8bf bhi = *(const v8bf*)(bp + k0 + 16);
    v16bf av = __builtin_shufflevector(alo, ahi, 0,1,2,3,4,5,6,7,8,9,10,11,12,13,14,15);
    v16bf bv = __builtin_shufflevector(blo, bhi, 0,1,2,3,4,5,6,7,8,9,10,11,12,13,14,15);
    acc = __builtin_amdgcn_wmma_f32_16x16x32_bf16(false, av, false, bv,
                                                  (short)0, acc, false, false);
  }

  const int n   = tn * 16 + lr;
  const float badd = bias ? bias[n] : 0.0f;
  #pragma unroll
  for (int v = 0; v < 8; ++v) {
    const int mr = tm * 16 + hi * 8 + v;      // C/D layout: VGPR v -> row v+8*hi
    float val = acc[v] + badd;
    if (flags & GF_RELU)  val = fmaxf(val, 0.0f);
    if (flags & GF_LEAKY) val = (val > 0.0f) ? val : 0.01f * val;
    const size_t o = (size_t)mr * ldc + n;
    if (flags & GF_RESID) val += Ci[o];
    Cc[o] = val;
    if (Cb) Cb[o] = (__bf16)val;
  }
}

// --------------------------- weight prep kernels ---------------------------
// Wt[n*K+k] = (bf16) W[k*N+n]   (transpose + downconvert, write-coalesced)
__global__ void conv_transpose_k(const float* __restrict__ W, __bf16* __restrict__ Wt,
                                 int K, int N)
{
  const int idx = blockIdx.x * 256 + threadIdx.x;
  if (idx >= K * N) return;
  const int n = idx / K, k = idx - n * K;
  Wt[idx] = (__bf16)W[(size_t)k * N + n];
}

// Wout[i,j,zc] -> P[(zc*64+j)*64 + i]  (so OPM stage-2 B^T rows are contiguous)
__global__ void perm_wout_k(const float* __restrict__ W, __bf16* __restrict__ P)
{
  const int idx = blockIdx.x * 256 + threadIdx.x;
  if (idx >= 64 * 64 * 64) return;
  const int i  = idx & 63;
  const int j  = (idx >> 6) & 63;
  const int zc = idx >> 12;
  P[idx] = (__bf16)W[(size_t)i * 4096 + j * 64 + zc];
}

// --------------------------- small pipeline kernels ------------------------
__global__ void gather_emb_k(float* __restrict__ eg, const float* __restrict__ emb,
                             const int* __restrict__ ids)
{
  const int idx = blockIdx.x * 256 + threadIdx.x;
  if (idx >= kR * kCM) return;
  const int r = idx >> 9, c = idx & 511;
  eg[idx] = emb[(size_t)ids[r] * kCM + c];
}

// ContinuousValueEncoder stage 1: h = relu(clip(v,512)*cw1 + cb1), bf16 out
__global__ void cve1_k(__bf16* __restrict__ h, const float* __restrict__ values,
                       const float* __restrict__ cw1, const float* __restrict__ cb1)
{
  const int idx = blockIdx.x * 256 + threadIdx.x;
  if (idx >= kR * kCM) return;
  const int r = idx >> 9, c = idx & 511;
  const float v = fminf(values[r], 512.0f);
  h[idx] = (__bf16)fmaxf(v * cw1[c] + cb1[c], 0.0f);
}

// LayerNorm over rows of width 512; optional f32 write/accumulate + bf16 write
__global__ __launch_bounds__(256)
void ln512_k(const float* __restrict__ X, const float* __restrict__ g,
             const float* __restrict__ b, float* __restrict__ Yf,
             __bf16* __restrict__ Yb, int addFlag)
{
  __shared__ float red[256];
  const int row = blockIdx.x, tid = threadIdx.x;
  const float* x = X + (size_t)row * kCM;
  float xs[2];
  float s = 0.f;
  #pragma unroll
  for (int i = 0; i < 2; ++i) { xs[i] = x[tid + i * 256]; s += xs[i]; }
  red[tid] = s; __syncthreads();
  for (int off = 128; off > 0; off >>= 1) {
    if (tid < off) red[tid] += red[tid + off];
    __syncthreads();
  }
  const float mu = red[0] * (1.0f / kCM);
  __syncthreads();
  s = 0.f;
  #pragma unroll
  for (int i = 0; i < 2; ++i) { const float d = xs[i] - mu; s += d * d; }
  red[tid] = s; __syncthreads();
  for (int off = 128; off > 0; off >>= 1) {
    if (tid < off) red[tid] += red[tid + off];
    __syncthreads();
  }
  const float rstd = rsqrtf(red[0] * (1.0f / kCM) + 1e-5f);
  #pragma unroll
  for (int i = 0; i < 2; ++i) {
    const int c = tid + i * 256;
    const float y = (xs[i] - mu) * rstd * g[c] + b[c];
    const size_t o = (size_t)row * kCM + c;
    if (Yf) { if (addFlag) Yf[o] += y; else Yf[o] = y; }
    if (Yb) Yb[o] = (__bf16)y;
  }
}

// Fused pair-bias + attention logits:
//   S[h,r,s] = (q[r,h,:]·k[s,h,:]) / 8  +  LN(z[r,s,:])·Wz[:,h]
// LN folded into precomputed gW/gWs/bW so z is read exactly once.
__global__ __launch_bounds__(256)
void scores_k(const float* __restrict__ z, const float* __restrict__ zg,
              const float* __restrict__ zb, const float* __restrict__ Wz,
              const float* __restrict__ q, const float* __restrict__ k,
              float* __restrict__ S)
{
  __shared__ float gW[kCZ * kNH];
  __shared__ float gWs[kNH];
  __shared__ float bW[kNH];
  const int tid = threadIdx.x;
  for (int t = tid; t < kCZ * kNH; t += 256) {
    const int c = t >> 3, h = t & 7;
    gW[t] = zg[c] * Wz[c * kNH + h];
  }
  if (tid < kNH) {
    float a = 0.f, bb = 0.f;
    for (int c = 0; c < kCZ; ++c) {
      a  += Wz[c * kNH + tid] * zg[c];
      bb += Wz[c * kNH + tid] * zb[c];
    }
    gWs[tid] = a; bW[tid] = bb;
  }
  __syncthreads();

  const size_t idx = (size_t)blockIdx.x * 256 + tid;   // over r*s = 1M pairs
  const int r = (int)(idx >> 10);
  const int s = (int)(idx & 1023);

  float zr[kCZ];
  const float4* zp = (const float4*)(z + idx * kCZ);
  float mu = 0.f;
  #pragma unroll
  for (int i = 0; i < 16; ++i) {
    const float4 t4 = zp[i];
    zr[4*i] = t4.x; zr[4*i+1] = t4.y; zr[4*i+2] = t4.z; zr[4*i+3] = t4.w;
    mu += t4.x + t4.y + t4.z + t4.w;
  }
  mu *= (1.0f / kCZ);
  float var = 0.f;
  #pragma unroll
  for (int c = 0; c < kCZ; ++c) { const float d = zr[c] - mu; var += d * d; }
  const float rstd = rsqrtf(var * (1.0f / kCZ) + 1e-5f);

  const float* qr = q + (size_t)r * kCH;
  const float* kr = k + (size_t)s * kCH;
  for (int h = 0; h < kNH; ++h) {
    float acc = 0.f;
    #pragma unroll
    for (int c = 0; c < kCZ; ++c) acc += zr[c] * gW[c * kNH + h];
    const float bias = rstd * (acc - mu * gWs[h]) + bW[h];
    float dot = 0.f;
    #pragma unroll
    for (int d = 0; d < kDH; ++d) dot += qr[h * kDH + d] * kr[h * kDH + d];
    S[((size_t)h << 20) + idx] = dot * 0.125f + bias;   // 1/sqrt(c_hidden)=1/8
  }
}

// Row softmax over 1024 keys -> bf16 probabilities (WMMA A operand)
__global__ __launch_bounds__(256)
void softmax_k(const float* __restrict__ S, __bf16* __restrict__ P)
{
  __shared__ float red[256];
  const size_t row = blockIdx.x;               // (h,r) row, 8192 rows
  const float* x = S + row * 1024;
  __bf16*      y = P + row * 1024;
  const int tid = threadIdx.x;
  float xs[4];
  float mx = -3.4e38f;
  #pragma unroll
  for (int i = 0; i < 4; ++i) { xs[i] = x[tid + i * 256]; mx = fmaxf(mx, xs[i]); }
  red[tid] = mx; __syncthreads();
  for (int off = 128; off > 0; off >>= 1) {
    if (tid < off) red[tid] = fmaxf(red[tid], red[tid + off]);
    __syncthreads();
  }
  const float m = red[0]; __syncthreads();
  float s = 0.f;
  #pragma unroll
  for (int i = 0; i < 4; ++i) { xs[i] = __expf(xs[i] - m); s += xs[i]; }
  red[tid] = s; __syncthreads();
  for (int off = 128; off > 0; off >>= 1) {
    if (tid < off) red[tid] += red[tid + off];
    __syncthreads();
  }
  const float inv = 1.0f / red[0];
  #pragma unroll
  for (int i = 0; i < 4; ++i) y[tid + i * 256] = (__bf16)(xs[i] * inv);
}

// V [s, h*8+d] -> Vt [h][16][s] bf16, d padded 8..15 with zeros (N=16 WMMA tile)
__global__ void make_vt_k(__bf16* __restrict__ Vt, const float* __restrict__ v)
{
  const int idx = blockIdx.x * 256 + threadIdx.x;
  if (idx >= kNH * 16 * kR) return;
  const int s = idx & 1023;
  const int d = (idx >> 10) & 15;
  const int h = idx >> 14;
  Vt[idx] = (d < kDH) ? (__bf16)v[(size_t)s * kCH + h * kDH + d] : (__bf16)0.0f;
}

// Obuf [h][r][16] -> ocat [r, h*8+d] bf16 (attention head merge)
__global__ void merge_o_k(__bf16* __restrict__ ocat, const float* __restrict__ Obuf)
{
  const int idx = blockIdx.x * 256 + threadIdx.x;
  if (idx >= kR * kCH) return;
  const int r = idx >> 6;
  const int h = (idx >> 3) & 7;
  const int d = idx & 7;
  ocat[idx] = (__bf16)Obuf[((size_t)h * kR + r) * 16 + d];
}

// pred[r] = d2[r,:]·dW3 + db3
__global__ __launch_bounds__(256)
void final_matvec_k(float* __restrict__ out, const float* __restrict__ X,
                    const float* __restrict__ w, const float* __restrict__ b)
{
  __shared__ float red[256];
  const int row = blockIdx.x, tid = threadIdx.x;
  const float* x = X + (size_t)row * kCM;
  red[tid] = x[tid] * w[tid] + x[tid + 256] * w[tid + 256];
  __syncthreads();
  for (int off = 128; off > 0; off >>= 1) {
    if (tid < off) red[tid] += red[tid + off];
    __syncthreads();
  }
  if (tid == 0) out[row] = red[0] + b[0];
}

// ---------------------------------------------------------------------------
extern "C" void kernel_launch(void* const* d_in, const int* in_sizes, int n_in,
                              void* d_out, int out_size, void* d_ws, size_t ws_size,
                              hipStream_t stream)
{
  (void)in_sizes; (void)n_in; (void)out_size;

  // ---- inputs (setup_inputs() dict order, params flattened in insertion order)
  const int*   gene_ids = (const int*)  d_in[0];
  const float* values   = (const float*)d_in[1];
  const float* z_in     = (const float*)d_in[2];
  const float* gene_emb = (const float*)d_in[3];
  const float* gln_g = (const float*)d_in[4],  *gln_b = (const float*)d_in[5];
  const float* cw1   = (const float*)d_in[6],  *cb1   = (const float*)d_in[7];
  const float* cw2   = (const float*)d_in[8],  *cb2   = (const float*)d_in[9];
  const float* cln_g = (const float*)d_in[10], *cln_b = (const float*)d_in[11];
  const float* aln_g = (const float*)d_in[12], *aln_b = (const float*)d_in[13];
  const float* zln_g = (const float*)d_in[14], *zln_b = (const float*)d_in[15];
  const float* Wz  = (const float*)d_in[16];
  const float* Wq  = (const float*)d_in[17];
  const float* Wk  = (const float*)d_in[18];
  const float* Wv  = (const float*)d_in[19];
  const float* Wo  = (const float*)d_in[20];
  const float* bo_ = (const float*)d_in[21];
  const float* tln_g = (const float*)d_in[22], *tln_b = (const float*)d_in[23];
  const float* Wt1 = (const float*)d_in[24], *bt1 = (const float*)d_in[25];
  const float* Wt2 = (const float*)d_in[26], *bt2 = (const float*)d_in[27];
  const float* oln_g = (const float*)d_in[28], *oln_b = (const float*)d_in[29];
  const float* Wp1 = (const float*)d_in[30], *bp1 = (const float*)d_in[31];
  const float* Wp2 = (const float*)d_in[32], *bp2 = (const float*)d_in[33];
  const float* Wout = (const float*)d_in[34], *bout = (const float*)d_in[35];
  const float* dW1 = (const float*)d_in[36], *db1 = (const float*)d_in[37];
  const float* dW2 = (const float*)d_in[38], *db2 = (const float*)d_in[39];
  const float* dW3 = (const float*)d_in[40], *db3 = (const float*)d_in[41];
  float* pred = (float*)d_out;

  // ---- workspace carve-up (256B aligned)
  char* cur = (char*)d_ws;
  auto alloc = [&](size_t bytes) -> void* {
    void* p = (void*)cur;
    cur += (bytes + 255) & ~(size_t)255;
    return p;
  };
  float*  z_work = (float*) alloc((size_t)kR * kR * kCZ * 4);      // 256 MB
  float*  S      = (float*) alloc((size_t)kNH * kR * kR * 4);      //  32 MB
  __bf16* Abf    = (__bf16*)alloc((size_t)kNH * kR * kR * 2);      //  16 MB
  float*  T      = (float*) alloc((size_t)kR * 4096 * 4);          //  16 MB
  __bf16* Tbf    = (__bf16*)alloc((size_t)kR * 4096 * 2);          //   8 MB
  float*  u      = (float*) alloc((size_t)kR * kFF * 4);
  __bf16* ubf    = (__bf16*)alloc((size_t)kR * kFF * 2);
  float*  m      = (float*) alloc((size_t)kR * kCM * 4);
  __bf16* mbf    = (__bf16*)alloc((size_t)kR * kCM * 2);
  float*  xbuf   = (float*) alloc((size_t)kR * kCM * 4);
  float*  eg     = (float*) alloc((size_t)kR * kCM * 4);
  __bf16* hbf    = (__bf16*)alloc((size_t)kR * kCM * 2);
  __bf16* mnbf   = (__bf16*)alloc((size_t)kR * kCM * 2);
  __bf16* tbf    = (__bf16*)alloc((size_t)kR * kCM * 2);
  __bf16* onbf   = (__bf16*)alloc((size_t)kR * kCM * 2);
  float*  qb     = (float*) alloc((size_t)kR * kCH * 4);
  float*  kb     = (float*) alloc((size_t)kR * kCH * 4);
  float*  vb     = (float*) alloc((size_t)kR * kCH * 4);
  __bf16* Vt     = (__bf16*)alloc((size_t)kNH * 16 * kR * 2);
  float*  Obuf   = (float*) alloc((size_t)kNH * kR * 16 * 4);
  __bf16* ocat   = (__bf16*)alloc((size_t)kR * kCH * 2);
  float*  ao     = (float*) alloc((size_t)kR * kCH * 4);
  __bf16* aobf   = (__bf16*)alloc((size_t)kR * kCH * 2);
  float*  bo2    = (float*) alloc((size_t)kR * kCH * 4);
  __bf16* bobf   = (__bf16*)alloc((size_t)kR * kCH * 2);
  float*  d1     = (float*) alloc((size_t)kR * kCM * 4);
  __bf16* d1bf   = (__bf16*)alloc((size_t)kR * kCM * 2);
  float*  d2     = (float*) alloc((size_t)kR * kCM * 4);
  // bf16 transposed weights
  __bf16* cw2t   = (__bf16*)alloc((size_t)kCM * kCM * 2);
  __bf16* WqT[2], *WkT[2], *WvT[2], *WoT[2], *Wt1T[2], *Wt2T[2];
  for (int b = 0; b < 2; ++b) {
    WqT[b]  = (__bf16*)alloc((size_t)kCH * kCM * 2);
    WkT[b]  = (__bf16*)alloc((size_t)kCH * kCM * 2);
    WvT[b]  = (__bf16*)alloc((size_t)kCH * kCM * 2);
    WoT[b]  = (__bf16*)alloc((size_t)kCM * kCH * 2);
    Wt1T[b] = (__bf16*)alloc((size_t)kFF * kCM * 2);
    Wt2T[b] = (__bf16*)alloc((size_t)kCM * kFF * 2);
  }
  __bf16* Wp1T  = (__bf16*)alloc((size_t)kCH * kCM * 2);   // OPM only needed b=0
  __bf16* Wp2T  = (__bf16*)alloc((size_t)kCH * kCM * 2);
  __bf16* WoutP = (__bf16*)alloc((size_t)4096 * kCH * 2);
  __bf16* dW1T  = (__bf16*)alloc((size_t)kCM * kCM * 2);
  __bf16* dW2T  = (__bf16*)alloc((size_t)kCM * kCM * 2);
  if ((size_t)(cur - (char*)d_ws) > ws_size) return;       // ws too small: bail

  auto nb = [](size_t n) { return (unsigned)((n + 255) / 256); };
  auto gemm = [&](const __bf16* A, long long sA, const __bf16* Bt, long long sB,
                  const float* bias, const float* Cin, long long sCin,
                  float* C, long long sC, __bf16* Cb,
                  int M, int N, int K, int lda, int ldb, int ldc,
                  int flags, int batch) {
    const int tiles = (M / 16) * (N / 16);
    dim3 g((unsigned)((tiles + 7) / 8), 1, (unsigned)batch);
    gemm_bf16_wmma<<<g, 256, 0, stream>>>(A, sA, Bt, sB, bias, Cin, sCin,
                                          C, sC, Cb, M, N, K, lda, ldb, ldc, flags);
  };
  auto tconv = [&](const float* W, __bf16* Wt, int K, int N) {
    conv_transpose_k<<<nb((size_t)K * N), 256, 0, stream>>>(W, Wt, K, N);
  };

  // ---- weight prep (bf16 + transpose; constant per launch)
  tconv(cw2, cw2t, kCM, kCM);
  for (int b = 0; b < 2; ++b) {
    tconv(Wq  + (size_t)b * kCM * kCH, WqT[b],  kCM, kCH);
    tconv(Wk  + (size_t)b * kCM * kCH, WkT[b],  kCM, kCH);
    tconv(Wv  + (size_t)b * kCM * kCH, WvT[b],  kCM, kCH);
    tconv(Wo  + (size_t)b * kCH * kCM, WoT[b],  kCH, kCM);
    tconv(Wt1 + (size_t)b * kCM * kFF, Wt1T[b], kCM, kFF);
    tconv(Wt2 + (size_t)b * kFF * kCM, Wt2T[b], kFF, kCM);
  }
  tconv(Wp1, Wp1T, kCM, kCH);       // block-0 OPM only (block-1 OPM is dead code)
  tconv(Wp2, Wp2T, kCM, kCH);
  perm_wout_k<<<nb(64 * 64 * 64), 256, 0, stream>>>(Wout, WoutP);
  tconv(dW1, dW1T, kCM, kCM);
  tconv(dW2, dW2T, kCM, kCM);

  // ---- embedding + continuous-value encoder
  gather_emb_k<<<nb((size_t)kR * kCM), 256, 0, stream>>>(eg, gene_emb, gene_ids);
  cve1_k<<<nb((size_t)kR * kCM), 256, 0, stream>>>(hbf, values, cw1, cb1);
  gemm(hbf, 0, cw2t, 0, cb2, nullptr, 0, xbuf, 0, nullptr,
       kR, kCM, kCM, kCM, kCM, kCM, 0, 1);
  ln512_k<<<kR, 256, 0, stream>>>(eg,   gln_g, gln_b, m, nullptr, 0);
  ln512_k<<<kR, 256, 0, stream>>>(xbuf, cln_g, cln_b, m, nullptr, 1);   // m += LN(x)

  // ---- 2 BioFormer blocks
  for (int b = 0; b < 2; ++b) {
    const float* zsrc = (b == 0) ? z_in : z_work;

    // attention input LN
    ln512_k<<<kR, 256, 0, stream>>>(m, aln_g + b * kCM, aln_b + b * kCM,
                                    nullptr, mnbf, 0);
    // q/k/v projections (no bias in reference)
    gemm(mnbf, 0, WqT[b], 0, nullptr, nullptr, 0, qb, 0, nullptr,
         kR, kCH, kCM, kCM, kCM, kCH, 0, 1);
    gemm(mnbf, 0, WkT[b], 0, nullptr, nullptr, 0, kb, 0, nullptr,
         kR, kCH, kCM, kCM, kCM, kCH, 0, 1);
    gemm(mnbf, 0, WvT[b], 0, nullptr, nullptr, 0, vb, 0, nullptr,
         kR, kCH, kCM, kCM, kCM, kCH, 0, 1);
    make_vt_k<<<nb((size_t)kNH * 16 * kR), 256, 0, stream>>>(Vt, vb);

    // fused z-LN + pair bias + qk logits, then softmax -> bf16 probs
    scores_k<<<nb((size_t)kR * kR), 256, 0, stream>>>(
        zsrc, zln_g + b * kCZ, zln_b + b * kCZ, Wz + b * kCZ * kNH, qb, kb, S);
    softmax_k<<<kNH * kR, 256, 0, stream>>>(S, Abf);

    // O[h] = A[h] @ V[h]   (batched over heads, N padded to 16)
    gemm(Abf, (long long)kR * kR, Vt, 16LL * kR, nullptr, nullptr, 0,
         Obuf, (long long)kR * 16, nullptr,
         kR, 16, kR, kR, kR, 16, 0, kNH);
    merge_o_k<<<nb((size_t)kR * kCH), 256, 0, stream>>>(ocat, Obuf);

    // m += o @ Wo + bo   (residual epilogue, in place)
    gemm(ocat, 0, WoT[b], 0, bo_ + b * kCM, m, 0, m, 0, nullptr,
         kR, kCM, kCH, kCH, kCH, kCM, GF_RESID, 1);

    // Transition FFN: m += relu(LN(m)@Wt1+bt1)@Wt2+bt2
    ln512_k<<<kR, 256, 0, stream>>>(m, tln_g + b * kCM, tln_b + b * kCM,
                                    nullptr, tbf, 0);
    gemm(tbf, 0, Wt1T[b], 0, bt1 + b * kFF, nullptr, 0, u, 0, ubf,
         kR, kFF, kCM, kCM, kCM, kFF, GF_RELU, 1);
    gemm(ubf, 0, Wt2T[b], 0, bt2 + b * kCM, m, 0, m, 0,
         (b == 1) ? mbf : nullptr,                 // final m also mirrored to bf16
         kR, kCM, kFF, kFF, kFF, kCM, GF_RESID, 1);

    // OuterProductMean: only block 0's z-update is live (block-1 z is never read)
    if (b == 0) {
      ln512_k<<<kR, 256, 0, stream>>>(m, oln_g, oln_b, nullptr, onbf, 0);
      gemm(onbf, 0, Wp1T, 0, bp1, nullptr, 0, ao, 0, aobf,
           kR, kCH, kCM, kCM, kCM, kCH, 0, 1);
      gemm(onbf, 0, Wp2T, 0, bp2, nullptr, 0, bo2, 0, bobf,
           kR, kCH, kCM, kCM, kCM, kCH, 0, 1);
      // T[r, zc*64+j] = sum_i ao[r,i] * Wout[i,j,zc]
      gemm(aobf, 0, WoutP, 0, nullptr, nullptr, 0, T, 0, Tbf,
           kR, 4096, kCH, kCH, kCH, 4096, 0, 1);
      // z_work[r,s,zc] = z_in[r,s,zc] + sum_j bo[s,j]*T[r,zc*64+j] + bout[zc]
      // batched per-r WMMA GEMM: M=s=1024, N=zc=64, K=j=64
      gemm(bobf, 0, Tbf, 4096, bout, z_in, (long long)kR * kCZ,
           z_work, (long long)kR * kCZ, nullptr,
           kR, kCZ, kCH, kCH, kCZ, kCZ, GF_RESID, kR);
    }
  }

  // ---- ExprDecoder
  gemm(mbf, 0, dW1T, 0, db1, nullptr, 0, d1, 0, d1bf,
       kR, kCM, kCM, kCM, kCM, kCM, GF_LEAKY, 1);
  gemm(d1bf, 0, dW2T, 0, db2, nullptr, 0, d2, 0, nullptr,
       kR, kCM, kCM, kCM, kCM, kCM, GF_LEAKY, 1);
  final_matvec_k<<<kR, 256, 0, stream>>>(pred, d2, dW3, db3);
}
